// BiPixelMambaLayer_34093450396623
// MI455X (gfx1250) — matrix operand order repaired and checked by reading, not compiled, not run on hardware
//
#include <hip/hip_runtime.h>

#define D_MODELc 96
#define D_STATEc 16
#define D_INNERc 192
#define DT_RANKc 6
#define NBATCHc  64
#define SEQ_L    1728
#define NTOK     (NBATCHc * SEQ_L)   /* 110592 */
#define EPSc     1e-5f

typedef __attribute__((ext_vector_type(16))) __bf16 v16bf;
typedef __attribute__((ext_vector_type(8)))  __bf16 bf16x8;
typedef __attribute__((ext_vector_type(8)))  float  v8f;

static __device__ __forceinline__ float sigmoidf_(float x) {
    return 1.0f / (1.0f + __expf(-x));
}

// CDNA5 async global->LDS copy (16B per lane), tracked by ASYNCcnt.
static __device__ __forceinline__ void async_copy_b128(unsigned lds_off, const void* gptr) {
    asm volatile("global_load_async_to_lds_b128 %0, %1, off"
                 :: "v"(lds_off), "v"(gptr) : "memory");
}
static __device__ __forceinline__ void async_wait0() {
    asm volatile("s_wait_asynccnt 0x0" ::: "memory");
}

// ---------------------------------------------------------------------------
// K1: pixel-shuffle gather + LayerNorm over C=96. One wave per token.
// Lane c holds channels c, c+32, c+64. Wave32 shfl_xor reduction.
// ---------------------------------------------------------------------------
__global__ __launch_bounds__(256)
void k_layernorm(const float* __restrict__ x, const float* __restrict__ nw,
                 const float* __restrict__ nbias, __bf16* __restrict__ xn)
{
    int gw   = blockIdx.x * 8 + (threadIdx.x >> 5);
    int lane = threadIdx.x & 31;
    if (gw >= NTOK) return;
    int nb = gw / SEQ_L, l = gw - nb * SEQ_L;
    int pz = nb >> 4, py = (nb >> 2) & 3, px = nb & 3;
    int nz = l / 144, ny = (l / 12) % 12, nx = l % 12;
    int z = nz * 4 + pz, h = ny * 4 + py, w = nx * 4 + px;

    float v[3];
#pragma unroll
    for (int i = 0; i < 3; ++i) {
        int c = lane + 32 * i;
        v[i] = x[((c * 48 + z) * 48 + h) * 48 + w];
    }
    float s  = v[0] + v[1] + v[2];
    float s2 = v[0] * v[0] + v[1] * v[1] + v[2] * v[2];
#pragma unroll
    for (int m = 16; m >= 1; m >>= 1) {
        s  += __shfl_xor(s,  m, 32);
        s2 += __shfl_xor(s2, m, 32);
    }
    float mu  = s * (1.0f / 96.0f);
    float var = s2 * (1.0f / 96.0f) - mu * mu;
    float inv = rsqrtf(var + EPSc);
#pragma unroll
    for (int i = 0; i < 3; ++i) {
        int c = lane + 32 * i;
        float o = (v[i] - mu) * inv * nw[c] + nbias[c];
        xn[gw * 96 + c] = (__bf16)o;
    }
}

// ---------------------------------------------------------------------------
// K0: fp32 -> bf16 weight convert with optional zero row padding.
// ---------------------------------------------------------------------------
__global__ __launch_bounds__(256)
void k_cvt_pad(const float* __restrict__ src, __bf16* __restrict__ dst,
               int src_rows, int K, int total)
{
    int i = blockIdx.x * 256 + threadIdx.x;
    if (i >= total) return;
    int r = i / K;
    dst[i] = (r < src_rows) ? (__bf16)src[i] : (__bf16)0.0f;
}

// ---------------------------------------------------------------------------
// K2: WMMA GEMM  out[M,N] = A[M,K] (bf16) * W[N,K]^T (bf16), fp32 accumulate.
// K = KTILES*32 (compile time). Block = 8 waves.
// Weights staged global->LDS once (CDNA5 async, ASYNCcnt); each wave loads
// ALL B fragments (NTILE x KTILES) into registers ONCE, then loops over
// MITER M-tiles. Per tile: clause-load all A fragments, one wait, then all
// NTILE*KTILES WMMAs back-to-back on register operands.
// MODE 0: store fp32 to outF (ldc = N).
// MODE 1: in_proj split (block-uniform: 64-col blocks never straddle 192):
//         cols <192 -> bf16 outX, cols >=192 -> bf16 outZ.
// reverse!=0: A row read at sequence-reversed position (backward stream).
// ---------------------------------------------------------------------------
template<int MODE, int NTILE, int KTILES, int MITER>
__global__ __launch_bounds__(256)
void k_wmma_gemm(const __bf16* __restrict__ A, const __bf16* __restrict__ W,
                 float* __restrict__ outF, __bf16* __restrict__ outX,
                 __bf16* __restrict__ outZ, int N, int reverse)
{
    constexpr int K = KTILES * 32;
    extern __shared__ __bf16 sW[];            // NTILE*16 rows x K bf16
    int tid  = threadIdx.x;
    int wave = tid >> 5;
    int lane = tid & 31;
    int half = lane >> 4;      // 0 or 1 (lane group)
    int l16  = lane & 15;
    int n0_base = blockIdx.y * (NTILE * 16);

    // ---- stage weights into LDS (async, 16B chunks) ----
    {
        const char* wsrc = (const char*)(W + (size_t)n0_base * K);
        unsigned    lbase = (unsigned)(size_t)sW;
        int chunks = NTILE * 2 * K;           // (NTILE*16*K*2)/16
        for (int i = tid; i < chunks; i += 256)
            async_copy_b128(lbase + i * 16, wsrc + i * 16);
        async_wait0();
    }
    __syncthreads();

    // ---- load all B fragments into registers (once per wave) ----
    v16bf breg[NTILE * KTILES];
#pragma unroll
    for (int j = 0; j < NTILE; ++j) {
#pragma unroll
        for (int kt = 0; kt < KTILES; ++kt) {
            const __bf16* brow = sW + (j * 16 + l16) * K + kt * 32 + half * 8;
            bf16x8 blo = *(const bf16x8*)(brow);
            bf16x8 bhi = *(const bf16x8*)(brow + 16);
            v16bf b;
#pragma unroll
            for (int e = 0; e < 8; ++e) { b[e] = blo[e]; b[e + 8] = bhi[e]; }
            breg[j * KTILES + kt] = b;
        }
    }

    // block-uniform output base for the in_proj split (never straddles 192)
    __bf16* outP = nullptr;
    if (MODE == 1)
        outP = (n0_base < 192) ? (outX + n0_base) : (outZ + (n0_base - 192));

    // ---- loop over M tiles, A streamed from global ----
#pragma unroll 1
    for (int it = 0; it < MITER; ++it) {
        int m0 = (blockIdx.x * (8 * MITER) + it * 8 + wave) * 16;

        int arow = m0 + l16;
        if (reverse) {
            int nb = arow / SEQ_L, l = arow - nb * SEQ_L;
            arow = nb * SEQ_L + (SEQ_L - 1 - l);
        }
        const __bf16* Arow = A + (long)arow * K;

        // clause-load ALL A fragments for this tile, then compute
        v16bf areg[KTILES];
#pragma unroll
        for (int kt = 0; kt < KTILES; ++kt) {
            int kb = kt * 32;
            // A 16x32 bf16: lane<16 K {kb..kb+7, kb+16..kb+23}, lane>=16 +8
            bf16x8 alo = *(const bf16x8*)(Arow + kb + half * 8);
            bf16x8 ahi = *(const bf16x8*)(Arow + kb + 16 + half * 8);
            v16bf a;
#pragma unroll
            for (int e = 0; e < 8; ++e) { a[e] = alo[e]; a[e + 8] = ahi[e]; }
            areg[kt] = a;
        }

        v8f acc[NTILE];
#pragma unroll
        for (int j = 0; j < NTILE; ++j) acc[j] = (v8f){};

#pragma unroll
        for (int kt = 0; kt < KTILES; ++kt) {
#pragma unroll
            for (int j = 0; j < NTILE; ++j) {
                acc[j] = __builtin_amdgcn_wmma_f32_16x16x32_bf16(
                             false, areg[kt], false, breg[j * KTILES + kt],
                             (short)0, acc[j], false, false);
            }
        }

        // epilogue: 32-bit index math, no divergent branches
        int rowbase = m0 + half * 8;
#pragma unroll
        for (int j = 0; j < NTILE; ++j) {
#pragma unroll
            for (int r = 0; r < 8; ++r) {
                int row = rowbase + r;
                if (MODE == 0) {
                    outF[row * N + (n0_base + j * 16 + l16)] = acc[j][r];
                } else {
                    outP[row * 192 + (j * 16 + l16)] = (__bf16)acc[j][r];
                }
            }
        }
    }
}

// ---------------------------------------------------------------------------
// K3: depthwise causal conv(4) + SiLU. Writes fp32 (scan) + bf16 (x_proj A).
// ---------------------------------------------------------------------------
__global__ __launch_bounds__(256)
void k_conv_silu(const __bf16* __restrict__ xpre, const float* __restrict__ cw,
                 const float* __restrict__ cb, float* __restrict__ xtf,
                 __bf16* __restrict__ xtb)
{
    int i = blockIdx.x * 256 + threadIdx.x;
    if (i >= NTOK * D_INNERc) return;
    int t = i / D_INNERc, d = i - t * D_INNERc;
    int nb = t / SEQ_L, l = t - nb * SEQ_L;
    float acc = cb[d];
#pragma unroll
    for (int j = 0; j < 4; ++j) {
        int ls = l - 3 + j;
        if (ls >= 0)
            acc += cw[d * 4 + j] * (float)xpre[(nb * SEQ_L + ls) * D_INNERc + d];
    }
    acc = acc * sigmoidf_(acc);   // SiLU
    xtf[i] = acc;
    xtb[i] = (__bf16)acc;
}

// ---------------------------------------------------------------------------
// K4: selective scan. One block per sequence (192 threads = channel d).
// 16 states per thread in registers. dtr/B/C are block-uniform loads
// (scalarizable to s_load broadcasts) -> no barriers in the serial loop.
// dt = softplus(dtr . dt_w[d] + dt_b[d]); A = -exp(A_log).
// ---------------------------------------------------------------------------
__global__ __launch_bounds__(192)
void k_scan(const float* __restrict__ dbl, const float* __restrict__ xtf,
            const float* __restrict__ dtw, const float* __restrict__ dtb,
            const float* __restrict__ Alog, const float* __restrict__ Dp,
            float* __restrict__ y)
{
    int nb = blockIdx.x;
    int d  = threadIdx.x;

    float Ar[16], h[16], wr[6];
#pragma unroll
    for (int n = 0; n < 16; ++n) {
        Ar[n] = -__expf(Alog[d * 16 + n]);
        h[n]  = 0.0f;
    }
#pragma unroll
    for (int r = 0; r < 6; ++r) wr[r] = dtw[d * 6 + r];
    float bias = dtb[d], Dd = Dp[d];

    for (int l = 0; l < SEQ_L; ++l) {
        long t = (long)nb * SEQ_L + l;
        const float* row = dbl + t * 48;      // block-uniform address
        float dtr = bias;
#pragma unroll
        for (int r = 0; r < 6; ++r) dtr += wr[r] * row[r];
        float dt = (dtr > 20.0f) ? dtr : __logf(1.0f + __expf(dtr));  // softplus
        float xv = xtf[t * 192 + d];
        float dx = dt * xv;
        float acc = 0.0f;
#pragma unroll
        for (int n = 0; n < 16; ++n) {
            float dA = __expf(dt * Ar[n]);
            h[n] = dA * h[n] + dx * row[6 + n];     // B_t
            acc += h[n] * row[22 + n];              // C_t
        }
        y[t * 192 + d] = acc + xv * Dd;
    }
}

// ---------------------------------------------------------------------------
// K5: gate + combine directions (backward read sequence-reversed) -> bf16 s.
// ---------------------------------------------------------------------------
__global__ __launch_bounds__(256)
void k_combine(const float* __restrict__ yf, const __bf16* __restrict__ zf,
               const float* __restrict__ yb, const __bf16* __restrict__ zb,
               __bf16* __restrict__ s)
{
    int i = blockIdx.x * 256 + threadIdx.x;
    if (i >= NTOK * D_INNERc) return;
    int t = i / D_INNERc, d = i - t * D_INNERc;
    int nb = t / SEQ_L, l = t - nb * SEQ_L;
    int tr = nb * SEQ_L + (SEQ_L - 1 - l);
    float z0 = (float)zf[t * 192 + d];
    float z1 = (float)zb[tr * 192 + d];
    float v = yf[t * 192 + d] * (z0 * sigmoidf_(z0))
            + yb[tr * 192 + d] * (z1 * sigmoidf_(z1));
    s[i] = (__bf16)v;
}

// ---------------------------------------------------------------------------
// K7: un-shuffle tokens back to (C,Z,H,W) + residual add.
// ---------------------------------------------------------------------------
__global__ __launch_bounds__(256)
void k_scatter(const float* __restrict__ tok, const float* __restrict__ x,
               float* __restrict__ out)
{
    const int VOL = 48 * 48 * 48;
    int i = blockIdx.x * 256 + threadIdx.x;
    if (i >= 96 * VOL) return;
    int c = i / VOL;
    int rem = i - c * VOL;
    int z = rem / (48 * 48), h = (rem / 48) % 48, w = rem % 48;
    int nz = z >> 2, pz = z & 3, ny = h >> 2, py = h & 3, nx = w >> 2, px = w & 3;
    int nb = (pz * 4 + py) * 4 + px;
    int l  = (nz * 12 + ny) * 12 + nx;
    int t = nb * SEQ_L + l;
    out[i] = tok[t * 96 + c] + x[i];
}

// ---------------------------------------------------------------------------
extern "C" void kernel_launch(void* const* d_in, const int* in_sizes, int n_in,
                              void* d_out, int out_size, void* d_ws, size_t ws_size,
                              hipStream_t stream)
{
    const float* x       = (const float*)d_in[0];
    const float* norm_w  = (const float*)d_in[1];
    const float* norm_b  = (const float*)d_in[2];
    const float* f_in_w  = (const float*)d_in[3];
    const float* f_cw    = (const float*)d_in[4];
    const float* f_cb    = (const float*)d_in[5];
    const float* f_xp    = (const float*)d_in[6];
    const float* f_dtw   = (const float*)d_in[7];
    const float* f_dtb   = (const float*)d_in[8];
    const float* f_Alog  = (const float*)d_in[9];
    const float* f_D     = (const float*)d_in[10];
    const float* b_in_w  = (const float*)d_in[11];
    const float* b_cw    = (const float*)d_in[12];
    const float* b_cb    = (const float*)d_in[13];
    const float* b_xp    = (const float*)d_in[14];
    const float* b_dtw   = (const float*)d_in[15];
    const float* b_dtb   = (const float*)d_in[16];
    const float* b_Alog  = (const float*)d_in[17];
    const float* b_D     = (const float*)d_in[18];
    const float* out_w   = (const float*)d_in[19];
    float* out = (float*)d_out;

    char* p = (char*)d_ws;
    auto alloc = [&](size_t bytes) -> void* {
        void* r = (void*)p;
        p += (bytes + 255) & ~(size_t)255;
        return r;
    };

    __bf16* xn    = (__bf16*)alloc((size_t)NTOK * 96 * 2);
    __bf16* wfin  = (__bf16*)alloc(384 * 96 * 2);
    __bf16* wbin  = (__bf16*)alloc(384 * 96 * 2);
    __bf16* wfxp  = (__bf16*)alloc(48 * 192 * 2);
    __bf16* wbxp  = (__bf16*)alloc(48 * 192 * 2);
    __bf16* wout  = (__bf16*)alloc(96 * 192 * 2);
    __bf16* xpreF = (__bf16*)alloc((size_t)NTOK * 192 * 2);
    __bf16* xpreB = (__bf16*)alloc((size_t)NTOK * 192 * 2);
    __bf16* zF    = (__bf16*)alloc((size_t)NTOK * 192 * 2);
    __bf16* zB    = (__bf16*)alloc((size_t)NTOK * 192 * 2);
    float*  xtfF  = (float*) alloc((size_t)NTOK * 192 * 4);
    float*  xtfB  = (float*) alloc((size_t)NTOK * 192 * 4);
    __bf16* xtbF  = (__bf16*)alloc((size_t)NTOK * 192 * 2);
    __bf16* xtbB  = (__bf16*)alloc((size_t)NTOK * 192 * 2);
    float*  dblF  = (float*) alloc((size_t)NTOK * 48 * 4);
    float*  dblB  = (float*) alloc((size_t)NTOK * 48 * 4);
    float*  yF    = (float*) alloc((size_t)NTOK * 192 * 4);
    float*  yB    = (float*) alloc((size_t)NTOK * 192 * 4);
    __bf16* sBuf  = (__bf16*)alloc((size_t)NTOK * 192 * 2);
    float*  tok   = (float*) alloc((size_t)NTOK * 96 * 4);

    // K1: layernorm (one wave per token)
    k_layernorm<<<NTOK / 8, 256, 0, stream>>>(x, norm_w, norm_b, xn);

    // K0: weight converts
    k_cvt_pad<<<(384 * 96 + 255) / 256, 256, 0, stream>>>(f_in_w, wfin, 384, 96, 384 * 96);
    k_cvt_pad<<<(384 * 96 + 255) / 256, 256, 0, stream>>>(b_in_w, wbin, 384, 96, 384 * 96);
    k_cvt_pad<<<(48 * 192 + 255) / 256, 256, 0, stream>>>(f_xp, wfxp, 38, 192, 48 * 192);
    k_cvt_pad<<<(48 * 192 + 255) / 256, 256, 0, stream>>>(b_xp, wbxp, 38, 192, 48 * 192);
    k_cvt_pad<<<(96 * 192 + 255) / 256, 256, 0, stream>>>(out_w, wout, 96, 192, 96 * 192);

    // 6912 M-tiles total; 8 waves x MITER=4 tiles per block -> 216 blocks_x
    const int MBLK = NTOK / 16 / (8 * 4);   // 216

    // K2: in_proj (forward + reversed-read backward), split into x / z
    // NTILE=4 (64 cols/block), KTILES=3 (K=96), grid.y=6, LDS = 12 KB
    k_wmma_gemm<1, 4, 3, 4><<<dim3(MBLK, 6), 256, 64 * 96 * 2, stream>>>(
        xn, wfin, nullptr, xpreF, zF, 384, 0);
    k_wmma_gemm<1, 4, 3, 4><<<dim3(MBLK, 6), 256, 64 * 96 * 2, stream>>>(
        xn, wbin, nullptr, xpreB, zB, 384, 1);

    // K3: depthwise conv + SiLU
    int nElem = NTOK * 192;
    k_conv_silu<<<(nElem + 255) / 256, 256, 0, stream>>>(xpreF, f_cw, f_cb, xtfF, xtbF);
    k_conv_silu<<<(nElem + 255) / 256, 256, 0, stream>>>(xpreB, b_cw, b_cb, xtfB, xtbB);

    // K2': x_proj (N padded 38 -> 48); NTILE=3, KTILES=6 (K=192), LDS = 18 KB
    k_wmma_gemm<0, 3, 6, 4><<<dim3(MBLK, 1), 256, 48 * 192 * 2, stream>>>(
        xtbF, wfxp, dblF, nullptr, nullptr, 48, 0);
    k_wmma_gemm<0, 3, 6, 4><<<dim3(MBLK, 1), 256, 48 * 192 * 2, stream>>>(
        xtbB, wbxp, dblB, nullptr, nullptr, 48, 0);

    // K4: selective scan per direction
    k_scan<<<NBATCHc, 192, 0, stream>>>(dblF, xtfF, f_dtw, f_dtb, f_Alog, f_D, yF);
    k_scan<<<NBATCHc, 192, 0, stream>>>(dblB, xtfB, b_dtw, b_dtb, b_Alog, b_D, yB);

    // K5: gate + combine (un-reverse backward)
    k_combine<<<(nElem + 255) / 256, 256, 0, stream>>>(yF, zF, yB, zB, sBuf);

    // K6: out_proj; NTILE=3, KTILES=6, grid.y=2 (N=96), LDS = 18 KB
    k_wmma_gemm<0, 3, 6, 4><<<dim3(MBLK, 2), 256, 48 * 192 * 2, stream>>>(
        sBuf, wout, tok, nullptr, nullptr, 96, 0);

    // K7: un-shuffle + residual
    k_scatter<<<(96 * 48 * 48 * 48 + 255) / 256, 256, 0, stream>>>(tok, x, out);
}
